// MultiHeadAttention_39728447488493
// MI455X (gfx1250) — matrix operand order repaired and checked by reading, not compiled, not run on hardware
//
#include <hip/hip_runtime.h>
#include <hip/hip_bf16.h>

// ---------------- problem constants ----------------
#define BB 2
#define SS 2048
#define DD 1024
#define HH 16
#define DH 64
#define TW 8   // waves per attention workgroup

typedef _Float16 half_t;
typedef __attribute__((ext_vector_type(16))) _Float16 v16h;
typedef __attribute__((ext_vector_type(8)))  _Float16 v8h;
typedef __attribute__((ext_vector_type(4)))  _Float16 v4h;
typedef __attribute__((ext_vector_type(8)))  float    v8f;
typedef __attribute__((ext_vector_type(4)))  float    v4f;

__device__ __forceinline__ v8f wmma_f16(v16h a, v16h b, v8f c) {
  return __builtin_amdgcn_wmma_f32_16x16x32_f16(
      false, a, false, b, (short)0, c, false, false);
}

// A fragment (16x32 f16) from a row-major half row (32-K chunk base):
// lane (m,g): elements 0..15 at K=16g..16g+15  -> two b128 loads.
__device__ __forceinline__ v16h load_a_frag(const half_t* row32, int g) {
  v8h lo = *(const v8h*)(row32 + 16 * g);
  v8h hi = *(const v8h*)(row32 + 16 * g + 8);
  return __builtin_shufflevector(lo, hi, 0, 1, 2, 3, 4, 5, 6, 7,
                                 8, 9, 10, 11, 12, 13, 14, 15);
}

__device__ __forceinline__ v16h cvt16(v4f x0, v4f x1, v4f x2, v4f x3) {
  v8h c01 = __builtin_shufflevector(__builtin_convertvector(x0, v4h),
                                    __builtin_convertvector(x1, v4h),
                                    0, 1, 2, 3, 4, 5, 6, 7);
  v8h c23 = __builtin_shufflevector(__builtin_convertvector(x2, v4h),
                                    __builtin_convertvector(x3, v4h),
                                    0, 1, 2, 3, 4, 5, 6, 7);
  return __builtin_shufflevector(c01, c23, 0, 1, 2, 3, 4, 5, 6, 7,
                                 8, 9, 10, 11, 12, 13, 14, 15);
}

// CDNA5 async global->LDS copy of 16 bytes per lane (ASYNCcnt-tracked).
__device__ __forceinline__ void async_copy_b128(const half_t* gsrc, half_t* ldst) {
  unsigned loff = (unsigned)(uintptr_t)ldst;   // low 32 bits = LDS offset
  asm volatile("global_load_async_to_lds_b128 %0, %1, off"
               :: "v"(loff), "v"(gsrc) : "memory");
}

// ------------------------------------------------------------------
// Kernel 1: per-head QKV projection.  One wave computes a 16x64 row tile
// (all four 16-col output tiles share the same A fragments).
//   which==0: Q -> qf  [bh][s][e]   (row major; A-frag friendly)
//   which==1: K -> kf  [bh][t][dh]  (row major; B-frag friendly for QK^T)
//   which==2: V -> vtf [bh][dh][t]  (transposed; B-frag friendly for P.V)
// ------------------------------------------------------------------
__global__ __launch_bounds__(32)
void mha_qkv_proj(const float* __restrict__ q, const float* __restrict__ k,
                  const float* __restrict__ v, const float* __restrict__ Wq,
                  const float* __restrict__ Wk, const float* __restrict__ Wv,
                  half_t* __restrict__ qf, half_t* __restrict__ kf,
                  half_t* __restrict__ vtf) {
  const int lane = threadIdx.x;
  const int m = lane & 15;
  const int g = lane >> 4;
  const int which = blockIdx.z;  // uniform
  const int bh = blockIdx.y;
  const int b = bh / HH, h = bh % HH;
  const int s0 = blockIdx.x * 16;

  const float* X = (which == 0) ? q : (which == 1) ? k : v;
  const float* W = (which == 0) ? Wq : (which == 1) ? Wk : Wv;
  const float* xrow  = X + ((size_t)b * SS + (s0 + m)) * DD + h * DH;
  const float* wbase = W + (size_t)h * DH * DH;

  // A fragments for both 32-K chunks, loaded once.
  v16h a[2];
#pragma unroll
  for (int kb2 = 0; kb2 < 2; ++kb2) {
    const float* p = xrow + kb2 * 32;
    a[kb2] = cvt16(*(const v4f*)(p + 16 * g), *(const v4f*)(p + 16 * g + 4),
                   *(const v4f*)(p + 16 * g + 8), *(const v4f*)(p + 16 * g + 12));
  }

  v8f c[4] = {};
#pragma unroll
  for (int et = 0; et < 4; ++et) {
    const float* wrow = wbase + (size_t)(et * 16 + m) * DH;
#pragma unroll
    for (int kb2 = 0; kb2 < 2; ++kb2) {
      const float* p = wrow + kb2 * 32 + 16 * g;
      v16h bm = cvt16(*(const v4f*)(p), *(const v4f*)(p + 4),
                      *(const v4f*)(p + 8), *(const v4f*)(p + 12));
      // NOTE: B uses K contiguous per lane, as A does -> cvt16 already gives
      // this order; the direct rebuild below is identical.
      v8h blo = __builtin_shufflevector(
          __builtin_convertvector(*(const v4f*)(p), v4h),
          __builtin_convertvector(*(const v4f*)(p + 4), v4h),
          0, 1, 2, 3, 4, 5, 6, 7);
      v8h bhi = __builtin_shufflevector(
          __builtin_convertvector(*(const v4f*)(p + 8), v4h),
          __builtin_convertvector(*(const v4f*)(p + 12), v4h),
          0, 1, 2, 3, 4, 5, 6, 7);
      bm = __builtin_shufflevector(blo, bhi, 0, 1, 2, 3, 4, 5, 6, 7,
                                   8, 9, 10, 11, 12, 13, 14, 15);
      c[et] = wmma_f16(a[kb2], bm, c[et]);
    }
  }

  if (which == 2) {
#pragma unroll
    for (int et = 0; et < 4; ++et) {
      v8h pv;
#pragma unroll
      for (int vi = 0; vi < 8; ++vi) pv[vi] = (half_t)c[et][vi];
      *(v8h*)(vtf + ((size_t)bh * DH + (et * 16 + m)) * SS + s0 + 8 * g) = pv;
    }
  } else {
    half_t* dst = ((which == 0) ? qf : kf) +
                  ((size_t)bh * SS + (s0 + 8 * g)) * DH + m;
#pragma unroll
    for (int et = 0; et < 4; ++et)
#pragma unroll
      for (int vi = 0; vi < 8; ++vi)
        dst[(size_t)vi * DH + et * 16] = (half_t)c[et][vi];
  }
}

// ------------------------------------------------------------------
// Kernel 2: flash attention. 8 waves/WG; each wave owns a 16-row Q tile;
// K (32x64) and V^T (64x32) tiles are double-buffered in LDS via
// global_load_async_to_lds_b128 and shared by all 8 waves.
// Row-sums computed by an extra WMMA against an all-ones B fragment.
// ------------------------------------------------------------------
__global__ __launch_bounds__(256)
void mha_attn(const half_t* __restrict__ qf, const half_t* __restrict__ kf,
              const half_t* __restrict__ vtf, half_t* __restrict__ concat) {
  __shared__ __align__(16) half_t kbuf[2][32 * 72];  // rows t_rel, stride 72
  __shared__ __align__(16) half_t vbuf[2][64 * 40];  // rows dh,    stride 40
  __shared__ __align__(16) half_t pbuf[TW][16 * 40]; // per-wave P tile

  const int tid = threadIdx.x;
  const int lane = tid & 31;
  const int w = tid >> 5;
  const int m = lane & 15;
  const int g = lane >> 4;
  const int bh = blockIdx.y;
  const int b = bh / HH, h = bh % HH;
  const int s0 = (blockIdx.x * TW + w) * 16;

  const half_t* kbase  = kf + (size_t)bh * SS * DH;   // [t][dh]
  const half_t* vtbase = vtf + (size_t)bh * DH * SS;  // [dh][t]

  // this thread's 16B copy slots: K 32x8 chunks, V^T 64x4 chunks (256 each)
  const int kr = tid >> 3, kc = tid & 7;
  const int vr = tid >> 2, vc = tid & 3;

  // issue tile 0 async copies
  async_copy_b128(kbase + (size_t)kr * DH + kc * 8, &kbuf[0][kr * 72 + kc * 8]);
  async_copy_b128(vtbase + (size_t)vr * SS + vc * 8, &vbuf[0][vr * 40 + vc * 8]);

  // Q A-fragments, 1/8 softmax scale folded (exact in f16)
  const half_t* qrow = qf + ((size_t)bh * SS + (s0 + m)) * DH;
  v16h aq0 = load_a_frag(qrow, g);
  v16h aq1 = load_a_frag(qrow + 32, g);
#pragma unroll
  for (int e = 0; e < 16; ++e) {
    aq0[e] = (half_t)((float)aq0[e] * 0.125f);
    aq1[e] = (half_t)((float)aq1[e] * 0.125f);
  }
  v16h ones;
#pragma unroll
  for (int e = 0; e < 16; ++e) ones[e] = (half_t)1.0f;

  float rowmax[8];
#pragma unroll
  for (int i = 0; i < 8; ++i) rowmax[i] = -3.0e38f;
  v8f acc[4] = {};
  v8f accsum = {};

  for (int t0 = 0; t0 < SS; t0 += 32) {
    const int buf = (t0 >> 5) & 1;
    if (t0 + 32 < SS) {   // prefetch next tile into the other buffer
      const int nb = buf ^ 1;
      async_copy_b128(kbase + (size_t)(t0 + 32 + kr) * DH + kc * 8,
                      &kbuf[nb][kr * 72 + kc * 8]);
      async_copy_b128(vtbase + (size_t)vr * SS + t0 + 32 + vc * 8,
                      &vbuf[nb][vr * 40 + vc * 8]);
      asm volatile("s_wait_asynccnt 0x2" ::: "memory"); // current tile landed
    } else {
      asm volatile("s_wait_asynccnt 0x0" ::: "memory");
    }
    __syncthreads();   // all waves' copies visible

    // ---- scores: two 16x16 tiles from LDS K ----
    v8f c0 = {}, c1 = {};
#pragma unroll
    for (int kb2 = 0; kb2 < 2; ++kb2) {
      v16h b0 = *(const v16h*)(&kbuf[buf][(size_t)m * 72 + kb2 * 32 + 16 * g]);
      v16h b1 = *(const v16h*)(&kbuf[buf][(size_t)(m + 16) * 72 + kb2 * 32 + 16 * g]);
      c0 = wmma_f16(kb2 ? aq1 : aq0, b0, c0);
      c1 = wmma_f16(kb2 ? aq1 : aq0, b1, c1);
    }

    // ---- online softmax: row max via 16-lane shuffles ----
    float p0[8], p1[8];
#pragma unroll
    for (int vi = 0; vi < 8; ++vi) {
      float t = fmaxf(c0[vi], c1[vi]);
      t = fmaxf(t, __shfl_xor(t, 1, 32));
      t = fmaxf(t, __shfl_xor(t, 2, 32));
      t = fmaxf(t, __shfl_xor(t, 4, 32));
      t = fmaxf(t, __shfl_xor(t, 8, 32));
      const float nm = fmaxf(rowmax[vi], t);
      const float alpha = __expf(rowmax[vi] - nm);
      rowmax[vi] = nm;
      p0[vi] = __expf(c0[vi] - nm);
      p1[vi] = __expf(c1[vi] - nm);
      accsum[vi] *= alpha;
#pragma unroll
      for (int j = 0; j < 4; ++j) acc[j][vi] *= alpha;
    }

    // ---- P: C layout -> LDS -> A layout ----
#pragma unroll
    for (int vi = 0; vi < 8; ++vi) {
      const int row = vi + 8 * g;
      pbuf[w][row * 40 + m]      = (half_t)p0[vi];
      pbuf[w][row * 40 + 16 + m] = (half_t)p1[vi];
    }
    asm volatile("s_wait_dscnt 0" ::: "memory");
    v8h plo = *(const v8h*)(&pbuf[w][m * 40 + 16 * g]);
    v8h phi = *(const v8h*)(&pbuf[w][m * 40 + 16 * g + 8]);
    v16h ap = __builtin_shufflevector(plo, phi, 0, 1, 2, 3, 4, 5, 6, 7,
                                      8, 9, 10, 11, 12, 13, 14, 15);

    // ---- row sums via WMMA against ones; O += P @ V from LDS V^T ----
    accsum = wmma_f16(ap, ones, accsum);
#pragma unroll
    for (int j = 0; j < 4; ++j) {
      v16h bv = *(const v16h*)(&vbuf[buf][(size_t)(j * 16 + m) * 40 + 16 * g]);
      acc[j] = wmma_f16(ap, bv, acc[j]);
    }
    __syncthreads();   // everyone done reading before next overwrite
  }

  // ---- normalize and store into concat [b][s][D] ----
#pragma unroll
  for (int vi = 0; vi < 8; ++vi) {
    const float inv = 1.0f / accsum[vi];
    const int s = s0 + vi + 8 * g;
#pragma unroll
    for (int j = 0; j < 4; ++j)
      concat[((size_t)b * SS + s) * DD + h * DH + j * 16 + m] =
          (half_t)(acc[j][vi] * inv);
  }
}

// ------------------------------------------------------------------
// Kernel 3: one-shot Wo f32 -> f16.
// ------------------------------------------------------------------
__global__ __launch_bounds__(256)
void mha_wo_cvt(const float* __restrict__ Wo, half_t* __restrict__ woh, int n) {
  const int i = blockIdx.x * blockDim.x + threadIdx.x;
  if (i < n) woh[i] = (half_t)Wo[i];
}

// ------------------------------------------------------------------
// Kernel 4: out = concat @ Wo^T + bo.  One wave computes a 32x64 tile
// (2 A-frags x 4 B-frags x 8 WMMAs per 32-K step) for B reuse.
// ------------------------------------------------------------------
__global__ __launch_bounds__(32)
void mha_out_proj(const half_t* __restrict__ concat,
                  const half_t* __restrict__ woh,
                  const float* __restrict__ bo, float* __restrict__ out) {
  const int lane = threadIdx.x;
  const int m = lane & 15;
  const int g = lane >> 4;
  const int ntiles = DD / 64;                 // 16
  const int r0 = (blockIdx.x / ntiles) * 32;  // rows in [0, B*S)
  const int j0 = (blockIdx.x % ntiles) * 64;  // cols in [0, D)

  v8f acc[2][4] = {};
  for (int kb = 0; kb < DD; kb += 32) {
    v16h a0 = load_a_frag(concat + (size_t)(r0 + m) * DD + kb, g);
    v16h a1 = load_a_frag(concat + (size_t)(r0 + 16 + m) * DD + kb, g);
#pragma unroll
    for (int nj = 0; nj < 4; ++nj) {
      v16h bm = *(const v16h*)(woh + (size_t)(j0 + nj * 16 + m) * DD + kb + 16 * g);
      acc[0][nj] = wmma_f16(a0, bm, acc[0][nj]);
      acc[1][nj] = wmma_f16(a1, bm, acc[1][nj]);
    }
  }

#pragma unroll
  for (int nj = 0; nj < 4; ++nj) {
    const float bias = bo[j0 + nj * 16 + m];
#pragma unroll
    for (int ri = 0; ri < 2; ++ri)
#pragma unroll
      for (int vi = 0; vi < 8; ++vi)
        out[(size_t)(r0 + ri * 16 + vi + 8 * g) * DD + j0 + nj * 16 + m] =
            acc[ri][nj][vi] + bias;
  }
}

// ------------------------------------------------------------------
extern "C" void kernel_launch(void* const* d_in, const int* in_sizes, int n_in,
                              void* d_out, int out_size, void* d_ws, size_t ws_size,
                              hipStream_t stream) {
  const float* q  = (const float*)d_in[0];
  const float* k  = (const float*)d_in[1];
  const float* v  = (const float*)d_in[2];
  const float* Wq = (const float*)d_in[3];
  const float* Wk = (const float*)d_in[4];
  const float* Wv = (const float*)d_in[5];
  const float* Wo = (const float*)d_in[6];
  const float* bo = (const float*)d_in[7];
  float* out = (float*)d_out;

  char* ws = (char*)d_ws;
  const size_t MB = (size_t)1 << 20;
  half_t* qf     = (half_t*)(ws);             // 8MB
  half_t* kf     = (half_t*)(ws + 8 * MB);    // 8MB
  half_t* vtf    = (half_t*)(ws + 16 * MB);   // 8MB
  half_t* concat = (half_t*)(ws + 24 * MB);   // 8MB
  half_t* woh    = (half_t*)(ws + 32 * MB);   // 2MB

  mha_wo_cvt<<<(DD * DD + 255) / 256, 256, 0, stream>>>(Wo, woh, DD * DD);

  dim3 gproj(SS / 16, BB * HH, 3);
  mha_qkv_proj<<<gproj, 32, 0, stream>>>(q, k, v, Wq, Wk, Wv, qf, kf, vtf);

  dim3 gattn(SS / (16 * TW), BB * HH);
  mha_attn<<<gattn, 256, 0, stream>>>(qf, kf, vtf, concat);

  mha_out_proj<<<(BB * SS / 32) * (DD / 64), 32, 0, stream>>>(concat, woh, bo, out);
}
